// MultiHeadAttention_15728170238257
// MI455X (gfx1250) — matrix operand order repaired and checked by reading, not compiled, not run on hardware
//
#include <hip/hip_runtime.h>
#include <hip/hip_bf16.h>
#include <math.h>

#define NB 16      // batch
#define NS 1024    // sequence length
#define ND 512     // model dim
#define NH 8       // heads
#define NDEP 64    // head depth
#define NEGV (-4294967295.0f)

typedef __attribute__((ext_vector_type(16))) __bf16 v16bf;
typedef __attribute__((ext_vector_type(8)))  float  v8f;
typedef __attribute__((ext_vector_type(4)))  unsigned int v4ui;
typedef __attribute__((ext_vector_type(8)))  int v8i;
typedef __attribute__((ext_vector_type(4)))  int v4i;
typedef unsigned short u16;

union bfvec { uint4 u[2]; v16bf v; };
union bfpk8 { uint4 u; __bf16 h[8]; };

// 16 contiguous bf16 -> v16bf (B-fragment: K-dim contiguous)
__device__ inline v16bf load16_contig(const __bf16* p) {
    bfvec t;
    t.u[0] = *(const uint4*)(p);
    t.u[1] = *(const uint4*)(p + 8);
    return t.v;
}

// A-fragment pattern: elements 0..7 from p[0..7], elements 8..15 from p[16..23]
__device__ inline v16bf load_split8(const __bf16* p) {
    bfvec t;
    t.u[0] = *(const uint4*)(p);
    t.u[1] = *(const uint4*)(p + 16);
    return t.v;
}

// ---- Tensor DMA descriptor builders (CDNA5 ISA) ----
// Group 0: [1:0]=count=1, [63:32]=lds_addr, [120:64]=global_addr, [127:126]=type=2
__device__ inline v4ui tdm_g0(unsigned lds_addr, unsigned long long gaddr) {
    v4ui g;
    g[0] = 1u;
    g[1] = lds_addr;
    g[2] = (unsigned)gaddr;
    g[3] = (unsigned)(gaddr >> 32) | (2u << 30);
    return g;
}
// Group 1 (2D tile): data_size=2B, tensor_dim0/1, tile_dim0/1, dim0 stride (48b)
__device__ inline v8i tdm_g1_2d(unsigned td0, unsigned td1,
                                unsigned tile0, unsigned tile1, unsigned stride0) {
    v8i g;
    g[0] = (int)(1u << 16);                                   // data_size = 2 bytes
    g[1] = (int)((td0 & 0xFFFFu) << 16);                      // tensor_dim0[15:0]
    g[2] = (int)((td0 >> 16) | ((td1 & 0xFFFFu) << 16));      // td0[31:16] | td1[15:0]
    g[3] = (int)((td1 >> 16) | (tile0 << 16));                // td1[31:16] | tile_dim0
    g[4] = (int)(tile1 & 0xFFFFu);                            // tile_dim1 (tile_dim2=0)
    g[5] = (int)stride0;                                      // tensor_dim0_stride[31:0]
    g[6] = 0;                                                 // stride0[47:32] | stride1 lo
    g[7] = 0;
    return g;
}

// ------------- weight prep: W[k][n] f32 -> Wt[which][n][k] bf16 (k-major) -------------
__global__ __launch_bounds__(256)
void prep_weights_kernel(const float* __restrict__ Wq, const float* __restrict__ Wk,
                         const float* __restrict__ Wv, u16* __restrict__ Wt)
{
    const int t = blockIdx.x * blockDim.x + threadIdx.x;   // 3 * 512 * 64 threads
    const int chunks = ND / 8;
    const int which = t / (ND * chunks);
    const int rem   = t % (ND * chunks);
    const int n     = rem / chunks;
    const int k0    = (rem % chunks) * 8;
    const float* W = (which == 0) ? Wq : (which == 1) ? Wk : Wv;
    bfpk8 pk;
    #pragma unroll
    for (int i = 0; i < 8; ++i) pk.h[i] = (__bf16)W[(size_t)(k0 + i) * ND + n];
    *(uint4*)((__bf16*)Wt + ((size_t)which * ND + n) * ND + k0) = pk.u;
}

// ------------- QKV projection: per-wave 16x64 tile; V written transposed -------------
__global__ __launch_bounds__(128)
void qkv_proj_kernel(const float* __restrict__ queries, const float* __restrict__ keys,
                     const u16* __restrict__ Wt,
                     const float* __restrict__ bq, const float* __restrict__ bk,
                     const float* __restrict__ bv,
                     u16* __restrict__ Qo, u16* __restrict__ Ko, u16* __restrict__ Vt)
{
    const int lane = threadIdx.x & 31;
    const int wave = threadIdx.x >> 5;
    const int hi = lane >> 4, lo = lane & 15;
    const int n0 = blockIdx.x << 6;
    const int m0 = ((blockIdx.y << 2) + wave) << 4;
    const int which = blockIdx.z;

    const float* X    = (which == 0) ? queries : keys;
    const float* bias = (which == 0) ? bq : (which == 1) ? bk : bv;
    const __bf16* W = (const __bf16*)Wt + (size_t)which * ND * ND;  // [n][k] bf16

    v8f acc[4] = {};
    const float* xrow = X + (size_t)(m0 + lo) * ND;

    for (int kk = 0; kk < ND; kk += 32) {
        v16bf a;
        {
            const float* xp = xrow + kk + (hi ? 16 : 0);
            float4 x0 = *(const float4*)(xp);
            float4 x1 = *(const float4*)(xp + 4);
            float4 x2 = *(const float4*)(xp + 8);
            float4 x3 = *(const float4*)(xp + 12);
            a[0]=(__bf16)x0.x;  a[1]=(__bf16)x0.y;  a[2]=(__bf16)x0.z;  a[3]=(__bf16)x0.w;
            a[4]=(__bf16)x1.x;  a[5]=(__bf16)x1.y;  a[6]=(__bf16)x1.z;  a[7]=(__bf16)x1.w;
            a[8]=(__bf16)x2.x;  a[9]=(__bf16)x2.y;  a[10]=(__bf16)x2.z; a[11]=(__bf16)x2.w;
            a[12]=(__bf16)x3.x; a[13]=(__bf16)x3.y; a[14]=(__bf16)x3.z; a[15]=(__bf16)x3.w;
        }
        const __bf16* wp = W + kk + (hi ? 16 : 0);
        #pragma unroll
        for (int g = 0; g < 4; ++g) {
            const v16bf bm = load16_contig(wp + (size_t)(n0 + g * 16 + lo) * ND);
            acc[g] = __builtin_amdgcn_wmma_f32_16x16x32_bf16(false, a, false, bm,
                                                             (short)0, acc[g], false, false);
        }
    }

    if (which < 2) {
        __bf16* Y = (__bf16*)((which == 0) ? Qo : Ko);
        #pragma unroll
        for (int g = 0; g < 4; ++g) {
            const int n = n0 + g * 16 + lo;
            const float bn = bias[n];
            #pragma unroll
            for (int j = 0; j < 8; ++j)
                Y[(size_t)(m0 + j + hi * 8) * ND + n] = (__bf16)(acc[g][j] + bn);
        }
    } else {
        __bf16* Y = (__bf16*)Vt;
        const int b  = m0 / NS;
        const int s0 = (m0 % NS) + hi * 8;
        #pragma unroll
        for (int g = 0; g < 4; ++g) {
            const int n = n0 + g * 16 + lo;
            const float bn = bias[n];
            bfpk8 pk;
            #pragma unroll
            for (int j = 0; j < 8; ++j) pk.h[j] = (__bf16)(acc[g][j] + bn);
            *(uint4*)(Y + ((size_t)b * ND + n) * NS + s0) = pk.u;
        }
    }
}

// ---------------- padding masks: sign(abs(rowsum)) ----------------
__global__ __launch_bounds__(256)
void mask_kernel(const float* __restrict__ queries, const float* __restrict__ keys,
                 float* __restrict__ qmask, float* __restrict__ kmask)
{
    const int r = blockIdx.x * blockDim.x + threadIdx.x;
    const float4* qp = (const float4*)(queries + (size_t)r * ND);
    const float4* kp = (const float4*)(keys    + (size_t)r * ND);
    float sq = 0.f, sk = 0.f;
    for (int i = 0; i < ND / 4; ++i) {
        float4 q = qp[i], k = kp[i];
        sq += q.x + q.y + q.z + q.w;
        sk += k.x + k.y + k.z + k.w;
    }
    qmask[r] = (fabsf(sq) != 0.f) ? 1.f : 0.f;
    kmask[r] = (fabsf(sk) != 0.f) ? 1.f : 0.f;
}

// ------- causal flash attention: 4 waves/block, TDM-staged K/V tiles in LDS -------
__global__ __launch_bounds__(128)
void flash_attn_kernel(const u16* __restrict__ Qraw, const u16* __restrict__ Kraw,
                       const u16* __restrict__ Vtraw,
                       const float* __restrict__ qmask, const float* __restrict__ kmask,
                       const float* __restrict__ queries, float* __restrict__ out)
{
    __shared__ __bf16 Ktile[32][64];     // 32 keys x 64 head dims (4 KB)
    __shared__ __bf16 Vtile[64][32];     // 64 head dims x 32 keys (4 KB)
    __shared__ __bf16 Plds[4][16][32];   // per-wave P tile for A-fragment transpose

    const int lane = threadIdx.x & 31;
    const int wave = threadIdx.x >> 5;
    const int hi = lane >> 4, lo = lane & 15;
    const int q0 = (blockIdx.x << 6) + (wave << 4);
    const int h  = blockIdx.y;
    const int b  = blockIdx.z;

    const __bf16* Qb  = (const __bf16*)Qraw;
    const __bf16* Kb  = (const __bf16*)Kraw;
    const __bf16* Vtb = (const __bf16*)Vtraw;

    const size_t headoff = (size_t)b * NS * ND + (size_t)h * NDEP;

    // Q A-fragments over head depth 64 (two K=32 fragments)
    const __bf16* qp = Qb + headoff + (size_t)(q0 + lo) * ND;
    const v16bf qf0 = load16_contig(qp + 0  + (hi ? 16 : 0));
    const v16bf qf1 = load16_contig(qp + 32 + (hi ? 16 : 0));

    v8f o[4] = {};
    float mi[8], li[8];
    #pragma unroll
    for (int j = 0; j < 8; ++j) { mi[j] = -INFINITY; li[j] = 0.f; }

    const __bf16* kbase = Kb + headoff;                                    // [s][d]
    const __bf16* vbase = Vtb + ((size_t)b * ND + (size_t)h * NDEP) * NS;  // [d][s]
    const float*  kmrow = kmask + (size_t)b * NS;

    // TDM descriptors: K tile = 64 contiguous dims x 32 key rows (row stride ND);
    //                  V tile = 32 contiguous keys x 64 dim rows (row stride NS).
    const v8i  g1k = tdm_g1_2d(ND, NS, 64, 32, ND);
    const v8i  g1v = tdm_g1_2d(NS, ND, 32, 64, NS);
    const v4i  z4  = {};
    const v8i  z8  = {};
    const unsigned kt_lds = (unsigned)(size_t)(void*)&Ktile[0][0];
    const unsigned vt_lds = (unsigned)(size_t)(void*)&Vtile[0][0];

    // Uniform trip count for the whole block (keeps barriers legal);
    // key blocks past a wave's causal range yield all-NEG scores -> exact no-op.
    const int kend = (blockIdx.x << 6) + 64;
    for (int kb = 0; kb < kend; kb += 32) {
        if (wave == 0) {
            const unsigned long long kga = (unsigned long long)(size_t)(kbase + (size_t)kb * ND);
            const unsigned long long vga = (unsigned long long)(size_t)(vbase + kb);
            __builtin_amdgcn_tensor_load_to_lds(tdm_g0(kt_lds, kga), g1k, z4, z4, z8, 0);
            __builtin_amdgcn_tensor_load_to_lds(tdm_g0(vt_lds, vga), g1v, z4, z4, z8, 0);
            __builtin_amdgcn_s_wait_tensorcnt(0);
        }
        __syncthreads();   // K/V tiles resident in LDS

        // K^T B-fragments from LDS: columns = keys (g*16+lo), contiguous k-dim
        const __bf16* kp0 = &Ktile[lo][hi ? 16 : 0];
        const __bf16* kp1 = &Ktile[16 + lo][hi ? 16 : 0];
        const v16bf kf00 = load16_contig(kp0);
        const v16bf kf01 = load16_contig(kp0 + 32);
        const v16bf kf10 = load16_contig(kp1);
        const v16bf kf11 = load16_contig(kp1 + 32);

        v8f sc0 = {}, sc1 = {};
        sc0 = __builtin_amdgcn_wmma_f32_16x16x32_bf16(false, qf0, false, kf00, (short)0, sc0, false, false);
        sc0 = __builtin_amdgcn_wmma_f32_16x16x32_bf16(false, qf1, false, kf01, (short)0, sc0, false, false);
        sc1 = __builtin_amdgcn_wmma_f32_16x16x32_bf16(false, qf0, false, kf10, (short)0, sc1, false, false);
        sc1 = __builtin_amdgcn_wmma_f32_16x16x32_bf16(false, qf1, false, kf11, (short)0, sc1, false, false);

        const float km0 = kmrow[kb + lo];
        const float km1 = kmrow[kb + 16 + lo];
        const int c0 = kb + lo, c1 = kb + 16 + lo;

        #pragma unroll
        for (int j = 0; j < 8; ++j) {
            const int row = q0 + j + hi * 8;
            float s0 = sc0[j] * 0.125f;          // 1/sqrt(64)
            float s1 = sc1[j] * 0.125f;
            if (c0 > row || km0 == 0.f) s0 = NEGV;
            if (c1 > row || km1 == 0.f) s1 = NEGV;

            float mx = fmaxf(s0, s1);
            mx = fmaxf(mx, __shfl_xor(mx, 1, 32));
            mx = fmaxf(mx, __shfl_xor(mx, 2, 32));
            mx = fmaxf(mx, __shfl_xor(mx, 4, 32));
            mx = fmaxf(mx, __shfl_xor(mx, 8, 32));

            const float mnew  = fmaxf(mi[j], mx);
            const float alpha = __expf(mi[j] - mnew);
            const float p0 = __expf(s0 - mnew);
            const float p1 = __expf(s1 - mnew);
            float rs = p0 + p1;
            rs += __shfl_xor(rs, 1, 32);
            rs += __shfl_xor(rs, 2, 32);
            rs += __shfl_xor(rs, 4, 32);
            rs += __shfl_xor(rs, 8, 32);
            li[j] = li[j] * alpha + rs;
            mi[j] = mnew;
            #pragma unroll
            for (int t = 0; t < 4; ++t) o[t][j] *= alpha;

            Plds[wave][j + hi * 8][lo]      = (__bf16)p0;
            Plds[wave][j + hi * 8][lo + 16] = (__bf16)p1;
        }
        __syncthreads();   // P tiles written
        const v16bf pf = load16_contig(&Plds[wave][lo][hi ? 16 : 0]);

        // V^T B-fragments from LDS: per lane fixed d-column, contiguous key rows
        #pragma unroll
        for (int t = 0; t < 4; ++t) {
            const v16bf vf = load16_contig(&Vtile[t * 16 + lo][hi ? 16 : 0]);
            o[t] = __builtin_amdgcn_wmma_f32_16x16x32_bf16(false, pf, false, vf,
                                                           (short)0, o[t], false, false);
        }
        __syncthreads();   // tiles fully consumed; safe to overwrite next iteration
    }

    const float* qmrow = qmask + (size_t)b * NS;
    #pragma unroll
    for (int j = 0; j < 8; ++j) {
        const int m = q0 + j + hi * 8;
        const float scale = qmrow[m] / li[j];
        const size_t base = (size_t)b * NS * ND + (size_t)m * ND + (size_t)h * NDEP + lo;
        out[base +  0] = o[0][j] * scale + queries[base +  0];
        out[base + 16] = o[1][j] * scale + queries[base + 16];
        out[base + 32] = o[2][j] * scale + queries[base + 32];
        out[base + 48] = o[3][j] * scale + queries[base + 48];
    }
}

extern "C" void kernel_launch(void* const* d_in, const int* in_sizes, int n_in,
                              void* d_out, int out_size, void* d_ws, size_t ws_size,
                              hipStream_t stream)
{
    const float* queries = (const float*)d_in[0];
    const float* keys    = (const float*)d_in[1];
    const float* Wq = (const float*)d_in[2];
    const float* bq = (const float*)d_in[3];
    const float* Wk = (const float*)d_in[4];
    const float* bk = (const float*)d_in[5];
    const float* Wv = (const float*)d_in[6];
    const float* bv = (const float*)d_in[7];
    float* out = (float*)d_out;

    const size_t nelem = (size_t)NB * NS * ND;
    u16* Qo = (u16*)d_ws;                          // bf16 Q activations [B*S][D]
    u16* Ko = Qo + nelem;                          // bf16 K activations [B*S][D]
    u16* Vt = Ko + nelem;                          // bf16 V transposed [B][D][S]
    u16* Wt = Vt + nelem;                          // bf16 weights transposed [3][N][K]
    float* kmask = (float*)(Wt + (size_t)3 * ND * ND);
    float* qmask = kmask + (size_t)NB * NS;

    prep_weights_kernel<<<(3 * ND * (ND / 8)) / 256, 256, 0, stream>>>(Wq, Wk, Wv, Wt);

    dim3 gProj(ND / 64, (NB * NS) / 64, 3);
    qkv_proj_kernel<<<gProj, dim3(128), 0, stream>>>(queries, keys, Wt, bq, bk, bv,
                                                     Qo, Ko, Vt);

    mask_kernel<<<(NB * NS) / 256, 256, 0, stream>>>(queries, keys, qmask, kmask);

    dim3 gAttn(NS / 64, NH, NB);
    flash_attn_kernel<<<gAttn, dim3(128), 0, stream>>>(Qo, Ko, Vt, qmask, kmask, queries, out);
}
